// VQEmbeddingEMA_52673478918650
// MI455X (gfx1250) — matrix operand order repaired and checked by reading, not compiled, not run on hardware
//
#include <hip/hip_runtime.h>
#include <hip/hip_bf16.h>

// ---------------- types ----------------
typedef __attribute__((ext_vector_type(8)))  float  v8f;
typedef __attribute__((ext_vector_type(16))) __bf16 v16bf;
typedef __attribute__((ext_vector_type(8)))  __bf16 bf16x8;
typedef __attribute__((ext_vector_type(4)))  __bf16 bf16x4;

#define B_DIM 16
#define T_DIM 4096
#define D_DIM 256
#define K_DIM 1024
#define ROWS  (B_DIM * T_DIM)         // 65536
#define QN    (B_DIM * D_DIM * T_DIM) // 16777216 quantized floats
#define LSTRIDE 280                   // bf16 elems per LDS row (16B aligned, conflict-free)
#define EPSF 1.1920928955078125e-07f

static __device__ __forceinline__ v16bf cat8(bf16x8 a, bf16x8 b) {
    return __builtin_shufflevector(a, b, 0,1,2,3,4,5,6,7,8,9,10,11,12,13,14,15);
}
static __device__ __forceinline__ v8f zero8() {
    v8f r;
    #pragma unroll
    for (int i = 0; i < 8; ++i) r[i] = 0.0f;
    return r;
}

// ---------------- prep: bf16 hi/lo split of codebook, |e_k|^2, zero histogram ----------------
__global__ __launch_bounds__(256) void vq_prep(const float* __restrict__ emb,
                                               __bf16* __restrict__ ehi,
                                               __bf16* __restrict__ elo,
                                               float*  __restrict__ nrm,
                                               unsigned int* __restrict__ counts) {
    const int k = blockIdx.x;       // 0..1023
    const int d = threadIdx.x;      // 0..255
    const float e = emb[k * D_DIM + d];
    const __bf16 h = (__bf16)e;
    const float  l = e - (float)h;
    ehi[k * D_DIM + d] = h;
    elo[k * D_DIM + d] = (__bf16)l;

    float s = e * e;
    #pragma unroll
    for (int m = 16; m >= 1; m >>= 1) s += __shfl_xor(s, m, 32);
    __shared__ float red[8];
    if ((threadIdx.x & 31) == 0) red[threadIdx.x >> 5] = s;
    __syncthreads();
    if (threadIdx.x == 0) {
        float t = 0.0f;
        #pragma unroll
        for (int i = 0; i < 8; ++i) t += red[i];
        nrm[k] = t;
        counts[k] = 0u;
    }
}

// ---------------- main: split-bf16 WMMA distance search + gather ----------------
__global__ __launch_bounds__(256, 1)
void vq_main(const float* __restrict__ x, const float* __restrict__ emb,
             const __bf16* __restrict__ ehi, const __bf16* __restrict__ elo,
             const float* __restrict__ nrm, unsigned int* __restrict__ counts,
             float* __restrict__ outQ, float* __restrict__ outIdx) {
    extern __shared__ char smemRaw[];
    __bf16* aHi = (__bf16*)smemRaw;                 // 8 waves * 16 rows * LSTRIDE
    __bf16* aLo = aHi + 8 * 16 * LSTRIDE;
    int*    wk  = (int*)(aLo + 8 * 16 * LSTRIDE);   // [8][16]

    const int tid  = threadIdx.x;
    const int lane = tid & 31;
    const int w    = tid >> 5;
    const int half = lane >> 4;
    const int l16  = lane & 15;

    const int rowBase = blockIdx.x * 128 + w * 16;  // grid = 512 blocks
    const int b  = rowBase >> 12;                   // / 4096
    const int t0 = rowBase & 4095;

    __bf16* myHi = aHi + w * (16 * LSTRIDE);
    __bf16* myLo = aLo + w * (16 * LSTRIDE);
    const float* xb = x + b * (D_DIM * T_DIM) + t0; // + d*T + tl

    // ---- stage this wave's 16x256 X-tile into LDS as bf16 hi/lo, [t][d] layout ----
    #pragma unroll 4
    for (int it = 0; it < 32; ++it) {
        const int q  = lane + (it << 5);   // 0..1023 quads
        const int tl = q & 15;
        const int d0 = (q >> 4) << 2;      // 0,4,...,252
        const float e0 = xb[(d0 + 0) * T_DIM + tl];
        const float e1 = xb[(d0 + 1) * T_DIM + tl];
        const float e2 = xb[(d0 + 2) * T_DIM + tl];
        const float e3 = xb[(d0 + 3) * T_DIM + tl];
        const __bf16 h0 = (__bf16)e0, h1 = (__bf16)e1, h2 = (__bf16)e2, h3 = (__bf16)e3;
        bf16x4 vh = {h0, h1, h2, h3};
        bf16x4 vl = {(__bf16)(e0 - (float)h0), (__bf16)(e1 - (float)h1),
                     (__bf16)(e2 - (float)h2), (__bf16)(e3 - (float)h3)};
        *(bf16x4*)(myHi + tl * LSTRIDE + d0) = vh;
        *(bf16x4*)(myLo + tl * LSTRIDE + d0) = vl;
    }
    __syncthreads();

    float bestS[8];
    int   bestK[8];
    #pragma unroll
    for (int v = 0; v < 8; ++v) { bestS[v] = -3.4e38f; bestK[v] = 0; }

    // ---- scan all 1024 codes in 8 groups of 128 (8 N-tiles of 16) ----
    for (int g = 0; g < 8; ++g) {
        v8f acc[8];
        #pragma unroll
        for (int nt = 0; nt < 8; ++nt) acc[nt] = zero8();

        #pragma unroll
        for (int dc8 = 0; dc8 < 8; ++dc8) {
            const int dc = dc8 * 32;
            // A fragment (16-bit A 16x32 layout): row = l16, K offs {half*8..+7, 16+half*8..+7}
            const __bf16* ah = myHi + l16 * LSTRIDE + dc + half * 8;
            const __bf16* al = myLo + l16 * LSTRIDE + dc + half * 8;
            const v16bf Ahi = cat8(*(const bf16x8*)ah, *(const bf16x8*)(ah + 16));
            const v16bf Alo = cat8(*(const bf16x8*)al, *(const bf16x8*)(al + 16));

            #pragma unroll
            for (int nt = 0; nt < 8; ++nt) {
                const int k = g * 128 + nt * 16 + l16;
                // B fragment (16-bit B 32x16 layout): col = l16, K offs {half*8..+7, 16+half*8..+7}
                const __bf16* bh = ehi + k * D_DIM + dc + half * 8;
                const __bf16* bl = elo + k * D_DIM + dc + half * 8;
                const v16bf Bhi = cat8(*(const bf16x8*)bh, *(const bf16x8*)(bh + 16));
                const v16bf Blo = cat8(*(const bf16x8*)bl, *(const bf16x8*)(bl + 16));
                acc[nt] = __builtin_amdgcn_wmma_f32_16x16x32_bf16(
                    false, Ahi, false, Bhi, (short)0, acc[nt], false, false);
                acc[nt] = __builtin_amdgcn_wmma_f32_16x16x32_bf16(
                    false, Ahi, false, Blo, (short)0, acc[nt], false, false);
                acc[nt] = __builtin_amdgcn_wmma_f32_16x16x32_bf16(
                    false, Alo, false, Bhi, (short)0, acc[nt], false, false);
            }
        }

        // ---- score: argmin dist == argmax (2*dot - |e_k|^2); ascending k keeps first-min ----
        #pragma unroll
        for (int nt = 0; nt < 8; ++nt) {
            const int k = g * 128 + nt * 16 + l16;
            const float nk = nrm[k];
            #pragma unroll
            for (int v = 0; v < 8; ++v) {
                const float s = 2.0f * acc[nt][v] - nk;
                if (s > bestS[v]) { bestS[v] = s; bestK[v] = k; }
            }
        }
    }

    // ---- reduce over the 16 lanes holding the same rows (xor bits 0..3) ----
    #pragma unroll
    for (int v = 0; v < 8; ++v) {
        float s = bestS[v];
        int   kk = bestK[v];
        #pragma unroll
        for (int m = 1; m <= 8; m <<= 1) {
            const float so = __shfl_xor(s, m, 32);
            const int   ko = __shfl_xor(kk, m, 32);
            if (so > s || (so == s && ko < kk)) { s = so; kk = ko; }
        }
        bestS[v] = s; bestK[v] = kk;
    }

    // lanes 0 (rows 0-7) and 16 (rows 8-15) publish indices + histogram
    if (l16 == 0) {
        #pragma unroll
        for (int v = 0; v < 8; ++v) {
            const int row = v + half * 8;
            const int kk = bestK[v];
            wk[w * 16 + row] = kk;
            outIdx[b * T_DIM + t0 + row] = (float)kk;
            atomicAdd(&counts[kk], 1u);
        }
    }
    __syncthreads();

    // ---- gather winning fp32 code rows -> quantized (B, 1, D, T) layout ----
    #pragma unroll 4
    for (int it = 0; it < 128; ++it) {
        const int e  = lane + (it << 5);   // 0..4095
        const int tl = e & 15;
        const int d  = e >> 4;
        const int kk = wk[w * 16 + tl];
        outQ[b * (D_DIM * T_DIM) + d * T_DIM + t0 + tl] = emb[kk * D_DIM + d];
    }
}

// ---------------- perplexity ----------------
__global__ __launch_bounds__(256) void vq_perp(const unsigned int* __restrict__ counts,
                                               float* __restrict__ outP) {
    __shared__ float red[256];
    float s = 0.0f;
    for (int k = threadIdx.x; k < K_DIM; k += 256) {
        const float p = (float)counts[k] * (1.0f / (float)ROWS);
        s += p * logf(p + EPSF);
    }
    red[threadIdx.x] = s;
    __syncthreads();
    for (int off = 128; off > 0; off >>= 1) {
        if (threadIdx.x < off) red[threadIdx.x] += red[threadIdx.x + off];
        __syncthreads();
    }
    if (threadIdx.x == 0) *outP = expf(-red[0]);
}

// ---------------- launcher ----------------
extern "C" void kernel_launch(void* const* d_in, const int* in_sizes, int n_in,
                              void* d_out, int out_size, void* d_ws, size_t ws_size,
                              hipStream_t stream) {
    const float* x   = (const float*)d_in[0];   // (16,1,256,4096)
    const float* emb = (const float*)d_in[1];   // (1,1024,256)

    char* ws = (char*)d_ws;
    __bf16* ehi = (__bf16*)(ws);                                  // 512 KB
    __bf16* elo = (__bf16*)(ws + 524288);                         // 512 KB
    float*  nrm = (float*)(ws + 1048576);                         // 4 KB
    unsigned int* counts = (unsigned int*)(ws + 1048576 + 4096);  // 4 KB

    float* outQ   = (float*)d_out;          // 16,777,216
    float* outIdx = outQ + QN;              // 65,536
    float* outP   = outIdx + ROWS;          // 1

    vq_prep<<<K_DIM, 256, 0, stream>>>(emb, ehi, elo, nrm, counts);

    const size_t smem = (size_t)2 * 8 * 16 * LSTRIDE * sizeof(__bf16) + 8 * 16 * sizeof(int);
    vq_main<<<ROWS / 128, 256, smem, stream>>>(x, emb, ehi, elo, nrm, counts, outQ, outIdx);

    vq_perp<<<1, 256, 0, stream>>>(counts, outP);
}
